// TrIP_20564303413650
// MI455X (gfx1250) — matrix operand order repaired and checked by reading, not compiled, not run on hardware
//
#include <hip/hip_runtime.h>
#include <hip/hip_bf16.h>
#include <math.h>

// ---------------------------------------------------------------------------
// TrIP forward for MI455X (gfx1250).  All per-edge GEMMs (radial MLPs, final
// MLP) run through v_wmma_f32_16x16x32_f16 with f16 operands / f32 accum.
// Weight matrices are swizzled once per launch into WMMA B-fragment order;
// each block stages its fragments into LDS once (ds_load_b128 feeds the
// WMMAs, 8x less L2 traffic).  The big R=h2@w3 tensors are never
// materialized: the conv_pair contraction is fused into the WMMA epilogue
// with f32 global atomics.
// ---------------------------------------------------------------------------

#define NE 65536          // edges
#define NN 4096           // nodes
#define NC 32             // channels C
#define NG 64             // graphs
#define CUTOFF 4.6f

typedef __attribute__((ext_vector_type(16))) _Float16 v16h;
typedef __attribute__((ext_vector_type(8)))  _Float16 v8h;
typedef __attribute__((ext_vector_type(8)))  float    v8f;

__device__ __forceinline__ unsigned f2ord(float f){
  unsigned u = __float_as_uint(f);
  return (u & 0x80000000u) ? ~u : (u | 0x80000000u);
}
__device__ __forceinline__ float ord2f(unsigned u){
  return __uint_as_float((u & 0x80000000u) ? (u ^ 0x80000000u) : ~u);
}

// ---------------- generic fill ----------------
__global__ void k_fill(float* p, float v, int n){
  int i = blockIdx.x*blockDim.x + threadIdx.x;
  if (i < n) p[i] = v;
}

__global__ void k_f32tof16(const float* __restrict__ s, _Float16* __restrict__ d, int n){
  int i = blockIdx.x*blockDim.x + threadIdx.x;
  if (i < n) d[i] = (_Float16)s[i];
}

// ---------------- weight pack into WMMA B-fragment order -------------------
// frag layout: [nt][kt][lane][16 halves]
//   lane l: n = nt*16 + (l&15);  halves h: k = kt*32 + (l>>4)*8 + (h&7) + (h>>3)*16
__global__ void k_pack(const float* __restrict__ W, _Float16* __restrict__ dst,
                       int K, int N, int KT){
  int idx = blockIdx.x*blockDim.x + threadIdx.x;
  int total = (N/16)*KT*32;
  if (idx >= total) return;
  int lane = idx & 31;
  int tile = idx >> 5;             // tile = nt*KT + kt
  int n  = (tile / KT)*16 + (lane & 15);
  int kb = (tile % KT)*32 + (lane >> 4)*8;
  _Float16* o = dst + ((size_t)tile*32 + lane)*16;
  #pragma unroll
  for (int h = 0; h < 16; h++){
    int k = kb + (h & 7) + ((h >> 3) << 4);
    o[h] = (_Float16)((k < K) ? W[(size_t)k*N + n] : 0.0f);
  }
}

// ---------------- WMMA GEMM with fused conv epilogue -----------------------
// One block = 8 waves = 8 consecutive 16-row tiles at one N-group; the
// block's B fragments are staged to LDS once.  M % 128 == 0 for all calls.
// mode 0: out16 = act(D+bias)        (f16 store, chains GEMMs)
// mode 1: out32 = act(D+bias)
// mode 2: acc[e*accLd + o] += D * xf[e*32 + i]            (n = o*32+i, nf=1)
// mode 3: acc[(e*16+o)*3+m] += D * t3[e*288+(i*3+f)*3+m]  (n = (o*32+i)*3+f)
template<int NT>
__global__ __launch_bounds__(256) void k_gemm(
    const _Float16* __restrict__ A, int lda,
    const _Float16* __restrict__ Bfrag, int KT,
    const float* __restrict__ bias,
    int M, int N, int act, int mode,
    _Float16* __restrict__ out16, float* __restrict__ out32,
    const float* __restrict__ xf, float* __restrict__ acc, int accLd,
    const float* __restrict__ t3)
{
  __shared__ _Float16 Bs[NT*2*512];        // KT <= 2 for every GEMM here
  int tid  = threadIdx.x;
  int lane = tid & 31;
  int wv   = tid >> 5;
  int tilesM  = M >> 4;
  int mblocks = tilesM >> 3;               // 8 M-tiles per block
  int mtb = (int)(blockIdx.x % mblocks);
  int ng  = (int)(blockIdx.x / mblocks);
  int mt  = mtb*8 + wv;
  int hi  = lane >> 4;
  int ln  = lane & 15;

  // cooperative stage of this block's B fragments into LDS (contiguous range)
  {
    int chunks = NT*KT*64;                 // 16-byte chunks
    const uint4* gsrc = (const uint4*)(Bfrag + (size_t)(ng*NT)*KT*32*16);
    uint4* ldst = (uint4*)Bs;
    for (int c = tid; c < chunks; c += 256) ldst[c] = gsrc[c];
  }
  __syncthreads();

  const v8f vzero = {};
  v8f accv[NT];
  #pragma unroll
  for (int t = 0; t < NT; t++) accv[t] = vzero;

  const _Float16* Arow = A + (size_t)(mt*16 + ln)*lda + hi*8;
  for (int kt = 0; kt < KT; kt++){
    // A fragment: halves 0..7 -> K = kt*32+hi*8+0..7 ; 8..15 -> +16
    v8h alo = *(const v8h*)(Arow + kt*32);
    v8h ahi = *(const v8h*)(Arow + kt*32 + 16);
    v16h a = __builtin_shufflevector(alo, ahi,
                 0,1,2,3,4,5,6,7,8,9,10,11,12,13,14,15);
    #pragma unroll
    for (int t = 0; t < NT; t++){
      v16h b = *(const v16h*)(Bs + ((size_t)(t*KT + kt)*32 + lane)*16);
      accv[t] = __builtin_amdgcn_wmma_f32_16x16x32_f16(
                    false, a, false, b, (short)0, accv[t], false, false);
    }
  }

  #pragma unroll
  for (int t = 0; t < NT; t++){
    int n = (ng*NT + t)*16 + ln;
    float bv = bias ? bias[n] : 0.0f;
    #pragma unroll
    for (int r = 0; r < 8; r++){
      int e = mt*16 + hi*8 + r;           // D layout: vgpr r, hi half -> M=r+8
      float v = accv[t][r] + bv;
      if (act == 1)      v = fmaxf(v, 0.0f);
      else if (act == 2) v = v * (1.0f/(1.0f + __expf(-v)));
      if (mode == 0)      out16[(size_t)e*N + n] = (_Float16)v;
      else if (mode == 1) out32[(size_t)e*N + n] = v;
      else if (mode == 2){
        int o = n >> 5, i = n & 31;
        atomicAdd(acc + (size_t)e*accLd + o, v * xf[(size_t)e*32 + i]);
      } else {
        int o = n/96; int rem = n - o*96; int i = rem/3; int f = rem - i*3;
        const float* tp = t3 + (size_t)e*288 + (i*3 + f)*3;
        float* ap = acc + ((size_t)e*16 + o)*3;
        atomicAdd(ap + 0, v*tp[0]);
        atomicAdd(ap + 1, v*tp[1]);
        atomicAdd(ap + 2, v*tp[2]);
      }
    }
  }
}

// ---------------- per-edge basis / radial features -------------------------
__global__ void k_edge_init(const float* __restrict__ rel,
                            _Float16* __restrict__ e0h,
                            float* __restrict__ b01, float* __restrict__ b11,
                            float* __restrict__ scaleA, float* __restrict__ escaleA){
  int e = blockIdx.x*blockDim.x + threadIdx.x;
  if (e >= NE) return;
  float rx = rel[e*3+0], ry = rel[e*3+1], rz = rel[e*3+2];
  float d  = sqrtf(rx*rx + ry*ry + rz*rz);
  float iv = 1.0f/d;
  float ux = rx*iv, uy = ry*iv, uz = rz*iv;
  float xr = d/CUTOFF;
  float sc = 1.0f - 1.0f/(1.0f - xr*xr);
  scaleA[e]  = sc;
  escaleA[e] = __expf(sc);

  const float dxc = CUTOFF/31.0f;
  _Float16* ep = e0h + (size_t)e*64;
  #pragma unroll
  for (int c = 0; c < 32; c++){
    float t = (d - c*dxc)/dxc;
    ep[c] = (_Float16)__expf(-2.0f*t*t);
  }
  ep[32] = (_Float16)d;
  #pragma unroll
  for (int c = 33; c < 64; c++) ep[c] = (_Float16)0.0f;

  const float SQ3 = 1.7320508075688772f;
  const float S15 = 3.872983346207417f;
  const float SQ5 = 2.23606797749979f;
  const float ISQ3 = 0.5773502691896258f;
  float Y1[3] = { SQ3*uy, SQ3*uz, SQ3*ux };
  b01[e*3+0] = uy; b01[e*3+1] = uz; b01[e*3+2] = ux;
  float Y2[5] = { S15*ux*uy, S15*uy*uz, 0.5f*SQ5*(3.0f*uz*uz - 1.0f),
                  S15*ux*uz, 0.5f*S15*(ux*ux - uy*uy) };

  float W1a[27]; float W2a[45];
  #pragma unroll
  for (int i = 0; i < 27; i++) W1a[i] = 0.0f;
  #pragma unroll
  for (int i = 0; i < 45; i++) W2a[i] = 0.0f;
  const float i6  = 0.4082482904638631f;     // 1/sqrt(6)
  W1a[0*9+1*3+2] =  i6; W1a[0*9+2*3+1] = -i6;
  W1a[1*9+2*3+0] =  i6; W1a[1*9+0*3+2] = -i6;
  W1a[2*9+0*3+1] =  i6; W1a[2*9+1*3+0] = -i6;
  const float i10 = 0.31622776601683794f;    // 1/sqrt(10)
  const float i30 = 0.18257418583505536f;    // 1/sqrt(30)
  W2a[0*9+2*3+0] = i10; W2a[0*9+0*3+2] = i10;
  W2a[1*9+0*3+1] = i10; W2a[1*9+1*3+0] = i10;
  W2a[2*9+0*3+0] = -i30; W2a[2*9+1*3+1] = 2.0f*i30; W2a[2*9+2*3+2] = -i30;
  W2a[3*9+2*3+1] = i10; W2a[3*9+1*3+2] = i10;
  W2a[4*9+2*3+2] = i10; W2a[4*9+0*3+0] = -i10;

  float* B = b11 + (size_t)e*27;            // [a(m)][b(n)][f]
  #pragma unroll
  for (int a = 0; a < 3; a++)
    #pragma unroll
    for (int b = 0; b < 3; b++){
      float j1 = 0.0f, j2 = 0.0f;
      #pragma unroll
      for (int k = 0; k < 3; k++) j1 += W1a[k*9 + a*3 + b]*Y1[k];
      #pragma unroll
      for (int k = 0; k < 5; k++) j2 += W2a[k*9 + a*3 + b]*Y2[k];
      B[(a*3+b)*3 + 0] = (a == b) ? ISQ3 : 0.0f;
      B[(a*3+b)*3 + 1] = j1;
      B[(a*3+b)*3 + 2] = j2;
    }
}

// ---------------- node embedding ----------------
__global__ void k_embed(const int* __restrict__ species, const float* __restrict__ emb,
                        float* __restrict__ f0){
  int i = blockIdx.x*blockDim.x + threadIdx.x;
  if (i >= NN*NC) return;
  int n = i >> 5, c = i & 31;
  f0[i] = emb[(size_t)(species[n] - 1)*NC + c];
}

// ---------------- per-edge gathers --------------------------------------
__global__ void k_gather_x0(const float* __restrict__ f0, const int* __restrict__ src,
                            float* __restrict__ xf0){
  int i = blockIdx.x*blockDim.x + threadIdx.x;
  if (i >= NE*NC) return;
  int e = i >> 5, c = i & 31;
  xf0[i] = f0[(size_t)src[e]*NC + c];
}

// t10[e,i] = sum_n x1[i,n]*b01[n]; t11[e,(i*3+f)*3+m] = sum_n x1[i,n]*b11[m][n][f]
__global__ void k_make_t(const float* __restrict__ f1, const int* __restrict__ src,
                         const float* __restrict__ b01, const float* __restrict__ b11,
                         float* __restrict__ t10, float* __restrict__ t11){
  int e = blockIdx.x*blockDim.x + threadIdx.x;
  if (e >= NE) return;
  const float* x1 = f1 + (size_t)src[e]*NC*3;   // [c][m]
  const float* bv = b01 + e*3;
  const float* B  = b11 + (size_t)e*27;
  for (int i = 0; i < NC; i++){
    float x0 = x1[i*3+0], xA = x1[i*3+1], xB = x1[i*3+2];
    t10[(size_t)e*NC + i] = x0*bv[0] + xA*bv[1] + xB*bv[2];
    #pragma unroll
    for (int f = 0; f < 3; f++)
      #pragma unroll
      for (int m = 0; m < 3; m++){
        float s = x0*B[(m*3+0)*3+f] + xA*B[(m*3+1)*3+f] + xB*B[(m*3+2)*3+f];
        t11[(size_t)e*288 + (i*3+f)*3 + m] = s;
      }
  }
}

// kv1[(e*co1+o)*3+m] += s01[e*co1+o]*b01[e,m]
__global__ void k_expand01(const float* __restrict__ s01, const float* __restrict__ b01,
                           float* __restrict__ kv1, int co1){
  int i = blockIdx.x*blockDim.x + threadIdx.x;
  if (i >= NE*co1) return;
  int e = i / co1;
  float s = s01[i];
  float* p = kv1 + (size_t)i*3;
  p[0] += s*b01[e*3+0];
  p[1] += s*b01[e*3+1];
  p[2] += s*b01[e*3+2];
}

// ---------------- attention --------------------------------------------
__global__ void k_node_q(const float* __restrict__ f0, const float* __restrict__ f1,
                         const float* __restrict__ wq0, const float* __restrict__ wq1,
                         float* __restrict__ q0, float* __restrict__ q1, int first){
  int n = blockIdx.x*blockDim.x + threadIdx.x;
  if (n >= NN) return;
  for (int o = 0; o < 8; o++){
    float s = 0.0f;
    for (int c = 0; c < NC; c++) s += wq0[o*NC + c]*f0[(size_t)n*NC + c];
    q0[n*8 + o] = s;
    if (!first){
      float s0 = 0, s1 = 0, s2 = 0;
      for (int c = 0; c < NC; c++){
        float w = wq1[o*NC + c];
        const float* f = f1 + ((size_t)n*NC + c)*3;
        s0 += w*f[0]; s1 += w*f[1]; s2 += w*f[2];
      }
      q1[(n*8 + o)*3 + 0] = s0;
      q1[(n*8 + o)*3 + 1] = s1;
      q1[(n*8 + o)*3 + 2] = s2;
    }
  }
}

__global__ void k_logits(const float* __restrict__ q0, const float* __restrict__ q1,
                         const float* __restrict__ kv0, const float* __restrict__ kv1,
                         const int* __restrict__ dst, const float* __restrict__ scaleA,
                         float* __restrict__ logits, int first){
  int e = blockIdx.x*blockDim.x + threadIdx.x;
  if (e >= NE) return;
  int d = dst[e];
  float sc = scaleA[e];
  float inv = first ? 1.0f : 0.5f;     // 1/sqrt(dk), dk = 1 or 4
  for (int h = 0; h < 8; h++){
    float l = q0[d*8 + h]*kv0[(size_t)e*16 + h];
    if (!first){
      const float* qq = q1 + (d*8 + h)*3;
      const float* kk = kv1 + ((size_t)e*16 + h)*3;
      l += qq[0]*kk[0] + qq[1]*kk[1] + qq[2]*kk[2];
    }
    logits[(size_t)e*8 + h] = l*inv + sc;
  }
}

__global__ void k_segmax(const float* __restrict__ logits, const int* __restrict__ dst,
                         unsigned* __restrict__ segm){
  int i = blockIdx.x*blockDim.x + threadIdx.x;
  if (i >= NE*8) return;
  int e = i >> 3, h = i & 7;
  atomicMax(segm + dst[e]*8 + h, f2ord(logits[i]));
}

__global__ void k_expsum(const float* __restrict__ logits, const int* __restrict__ dst,
                         const unsigned* __restrict__ segm,
                         float* __restrict__ ex, float* __restrict__ segs){
  int i = blockIdx.x*blockDim.x + threadIdx.x;
  if (i >= NE*8) return;
  int e = i >> 3, h = i & 7;
  float v = __expf(logits[i] - ord2f(segm[dst[e]*8 + h]));
  ex[i] = v;
  atomicAdd(segs + dst[e]*8 + h, v);
}

__global__ void k_agg(const float* __restrict__ ex, const float* __restrict__ segs,
                      const int* __restrict__ dst,
                      const float* __restrict__ kv0, const float* __restrict__ kv1,
                      float* __restrict__ z0, float* __restrict__ z1, int first){
  int i = blockIdx.x*blockDim.x + threadIdx.x;
  if (i >= NE*8) return;
  int e = i >> 3, h = i & 7;
  int d = dst[e];
  float att = ex[i]/segs[d*8 + h];
  atomicAdd(z0 + d*8 + h, att*kv0[(size_t)e*16 + 8 + h]);
  const float* v1 = first ? (kv1 + ((size_t)e*8 + h)*3)
                          : (kv1 + ((size_t)e*16 + 8 + h)*3);
  atomicAdd(z1 + (d*8 + h)*3 + 0, att*v1[0]);
  atomicAdd(z1 + (d*8 + h)*3 + 1, att*v1[1]);
  atomicAdd(z1 + (d*8 + h)*3 + 2, att*v1[2]);
}

__global__ void k_proj(const float* __restrict__ f0, const float* __restrict__ f1,
                       const float* __restrict__ z0, const float* __restrict__ z1,
                       const float* __restrict__ wp0, const float* __restrict__ wp1,
                       float* __restrict__ nf0, float* __restrict__ nf1, int first){
  int n = blockIdx.x*blockDim.x + threadIdx.x;
  if (n >= NN) return;
  for (int o = 0; o < NC; o++){
    float s = 0.0f;
    for (int c = 0; c < 8;  c++) s += wp0[o*40 + c]*z0[n*8 + c];
    for (int c = 0; c < NC; c++) s += wp0[o*40 + 8 + c]*f0[(size_t)n*NC + c];
    nf0[(size_t)n*NC + o] = s;
    float sm[3] = {0.f, 0.f, 0.f};
    if (first){
      for (int c = 0; c < 8; c++){
        float w = wp1[o*8 + c];
        const float* zz = z1 + (n*8 + c)*3;
        sm[0] += w*zz[0]; sm[1] += w*zz[1]; sm[2] += w*zz[2];
      }
    } else {
      for (int c = 0; c < 8; c++){
        float w = wp1[o*40 + c];
        const float* zz = z1 + (n*8 + c)*3;
        sm[0] += w*zz[0]; sm[1] += w*zz[1]; sm[2] += w*zz[2];
      }
      for (int c = 0; c < NC; c++){
        float w = wp1[o*40 + 8 + c];
        const float* ff = f1 + ((size_t)n*NC + c)*3;
        sm[0] += w*ff[0]; sm[1] += w*ff[1]; sm[2] += w*ff[2];
      }
    }
    float* po = nf1 + ((size_t)n*NC + o)*3;
    po[0] = sm[0]; po[1] = sm[1]; po[2] = sm[2];
  }
}

// ---------------- final readout ----------------
__global__ void k_final(const _Float16* __restrict__ hb, const float* __restrict__ w,
                        const float* __restrict__ b, const int* __restrict__ eg,
                        const float* __restrict__ esc, float* __restrict__ out){
  int e = blockIdx.x*blockDim.x + threadIdx.x;
  if (e >= NE) return;
  float s = b[0];
  for (int c = 0; c < 64; c++) s += (float)hb[(size_t)e*64 + c]*w[c];
  atomicAdd(out + eg[e], esc[e]*s);
}

// ===========================================================================
extern "C" void kernel_launch(void* const* d_in, const int* in_sizes, int n_in,
                              void* d_out, int out_size, void* d_ws, size_t ws_size,
                              hipStream_t stream){
  (void)in_sizes; (void)n_in; (void)out_size; (void)ws_size;
  const int*   species = (const int*)  d_in[0];
  const int*   src     = (const int*)  d_in[1];
  const int*   dst     = (const int*)  d_in[2];
  const int*   egraph  = (const int*)  d_in[3];
  const float* rel     = (const float*)d_in[4];

  // params flattened in jax pytree order (dict keys sorted alphabetically)
  int pi = 5;
  auto nf = [&]() -> const float* { return (const float*)d_in[pi++]; };
  struct RadP { const float *b1, *b2, *w1, *w2, *w3; };
  auto getRad = [&](){ RadP r; r.b1=nf(); r.b2=nf(); r.w1=nf(); r.w2=nf(); r.w3=nf(); return r; };

  const float* embedding = nf();
  RadP fr00 = getRad();          // final_radial '00'
  RadP fr10 = getRad();          // final_radial '10'
  RadP Lr[3][4];
  const float *Lwp0[3], *Lwp1[3], *Lwq0[3], *Lwq1[3];
  for (int L = 0; L < 3; L++){
    int nrad = (L == 0) ? 2 : 4;
    for (int r = 0; r < nrad; r++) Lr[L][r] = getRad();
    Lwp0[L] = nf(); Lwp1[L] = nf();
    Lwq0[L] = nf(); Lwq1[L] = (L == 0) ? nullptr : nf();
  }
  const float* mb0 = nf(); const float* mw0 = nf();
  const float* mb1 = nf(); const float* mw1 = nf();
  const float* mb2 = nf(); const float* mw2 = nf();

  // ---------------- workspace bump allocator ----------------
  char* base = (char*)d_ws;
  size_t off = 0;
  auto alloc = [&](size_t bytes) -> void* {
    off = (off + 255) & ~(size_t)255;
    void* p = base + off;
    off += bytes;
    return p;
  };

  _Float16* e0h    = (_Float16*)alloc((size_t)NE*64*2);
  float*    b01    = (float*)   alloc((size_t)NE*3*4);
  float*    b11    = (float*)   alloc((size_t)NE*27*4);
  float*    scaleA = (float*)   alloc((size_t)NE*4);
  float*    escale = (float*)   alloc((size_t)NE*4);
  float*    f0A    = (float*)   alloc((size_t)NN*NC*4);
  float*    f0B    = (float*)   alloc((size_t)NN*NC*4);
  float*    f1A    = (float*)   alloc((size_t)NN*NC*3*4);
  float*    f1B    = (float*)   alloc((size_t)NN*NC*3*4);
  float*    xf0    = (float*)   alloc((size_t)NE*NC*4);
  float*    t10    = (float*)   alloc((size_t)NE*NC*4);
  float*    t11    = (float*)   alloc((size_t)NE*288*4);
  _Float16* h1     = (_Float16*)alloc((size_t)NE*32*2);
  _Float16* h2     = (_Float16*)alloc((size_t)NE*32*2);
  float*    kv0    = (float*)   alloc((size_t)NE*16*4);
  float*    kv1    = (float*)   alloc((size_t)NE*16*3*4);
  float*    s01    = (float*)   alloc((size_t)NE*16*4);
  float*    q0     = (float*)   alloc((size_t)NN*8*4);
  float*    q1     = (float*)   alloc((size_t)NN*8*3*4);
  float*    logits = (float*)   alloc((size_t)NE*8*4);
  float*    exb    = (float*)   alloc((size_t)NE*8*4);
  unsigned* segm   = (unsigned*)alloc((size_t)NN*8*4);
  float*    segs   = (float*)   alloc((size_t)NN*8*4);
  float*    z0     = (float*)   alloc((size_t)NN*8*4);
  float*    z1     = (float*)   alloc((size_t)NN*8*3*4);
  float*    hbuf   = (float*)   alloc((size_t)NE*64*4);
  _Float16* hb16   = (_Float16*)alloc((size_t)NE*64*2);
  _Float16* hb16b  = (_Float16*)alloc((size_t)NE*64*2);

  auto fill = [&](float* p, float v, int n){
    k_fill<<<(n + 255)/256, 256, 0, stream>>>(p, v, n);
  };

  // pack one weight matrix into fragment order
  auto pack = [&](const float* W, int K, int N) -> const _Float16* {
    int KT = (K + 31)/32;
    _Float16* dstF = (_Float16*)alloc((size_t)(N/16)*KT*32*16*2);
    int total = (N/16)*KT*32;
    k_pack<<<(total + 255)/256, 256, 0, stream>>>(W, dstF, K, N, KT);
    return dstF;
  };

  auto gemm = [&](const _Float16* A, int lda, const _Float16* Bf, int KT,
                  const float* bias, int M, int N, int act, int mode,
                  _Float16* o16, float* o32,
                  const float* xf, float* acc, int accLd, const float* t3){
    if (N % 64 == 0){
      int blocks = (M/128)*(N/64);
      k_gemm<4><<<blocks, 256, 0, stream>>>(A, lda, Bf, KT, bias, M, N,
          act, mode, o16, o32, xf, acc, accLd, t3);
    } else {
      int blocks = (M/128)*(N/32);
      k_gemm<2><<<blocks, 256, 0, stream>>>(A, lda, Bf, KT, bias, M, N,
          act, mode, o16, o32, xf, acc, accLd, t3);
    }
  };

  struct RadF { const _Float16 *w1f, *w2f, *w3f; const float *b1, *b2; int N3; };
  auto packRad = [&](const RadP& r, int N3){
    RadF f;
    f.w1f = pack(r.w1, 33, 32);
    f.w2f = pack(r.w2, 32, 32);
    f.w3f = pack(r.w3, 32, N3);
    f.b1 = r.b1; f.b2 = r.b2; f.N3 = N3;
    return f;
  };
  // radial MLP: e0h -> h1 -> h2 (relu, f16) -> w3 with fused conv epilogue
  auto runRadial = [&](const RadF& f, int mode, const float* xf,
                       float* acc, int accLd, const float* t3){
    gemm(e0h, 64, f.w1f, 2, f.b1, NE, 32, 1, 0, h1, nullptr, nullptr, nullptr, 0, nullptr);
    gemm(h1,  32, f.w2f, 1, f.b2, NE, 32, 1, 0, h2, nullptr, nullptr, nullptr, 0, nullptr);
    gemm(h2,  32, f.w3f, 1, nullptr, NE, f.N3, 0, mode, nullptr, nullptr, xf, acc, accLd, t3);
  };

  // pack all weights (re-done each launch: deterministic, graph-capture safe)
  RadF F00 = packRad(fr00, 2048), F10 = packRad(fr10, 2048);
  RadF LF[3][4];
  LF[0][0] = packRad(Lr[0][0], 512);
  LF[0][1] = packRad(Lr[0][1], 256);
  for (int L = 1; L < 3; L++){
    LF[L][0] = packRad(Lr[L][0], 512);
    LF[L][1] = packRad(Lr[L][1], 512);
    LF[L][2] = packRad(Lr[L][2], 512);
    LF[L][3] = packRad(Lr[L][3], 1536);
  }
  const _Float16* mw0f = pack(mw0, 64, 64);
  const _Float16* mw1f = pack(mw1, 64, 64);

  // ---------------- stage 0: per-edge basis + embedding ----------------
  k_edge_init<<<NE/256, 256, 0, stream>>>(rel, e0h, b01, b11, scaleA, escale);
  k_embed<<<(NN*NC)/256, 256, 0, stream>>>(species, embedding, f0A);

  float* f0  = f0A; float* f0n = f0B;
  float* f1  = f1A; float* f1n = f1B;

  // ---------------- 3 attention layers ----------------
  for (int L = 0; L < 3; L++){
    int first = (L == 0);
    int co1 = first ? 8 : 16;

    k_gather_x0<<<(NE*NC)/256, 256, 0, stream>>>(f0, src, xf0);
    if (!first) k_make_t<<<NE/256, 256, 0, stream>>>(f1, src, b01, b11, t10, t11);

    fill(kv0, 0.0f, NE*16);
    fill(kv1, 0.0f, NE*co1*3);
    fill(s01, 0.0f, NE*co1);

    runRadial(LF[L][0], 2, xf0, kv0, 16, nullptr);            // '00' -> kv0
    runRadial(LF[L][1], 2, xf0, s01, co1, nullptr);           // '01' -> s01
    if (!first){
      runRadial(LF[L][2], 2, t10, kv0, 16, nullptr);          // '10' -> kv0
      runRadial(LF[L][3], 3, nullptr, kv1, 0, t11);           // '11' -> kv1
    }
    k_expand01<<<(NE*co1 + 255)/256, 256, 0, stream>>>(s01, b01, kv1, co1);

    k_node_q<<<(NN + 255)/256, 256, 0, stream>>>(f0, f1, Lwq0[L], Lwq1[L], q0, q1, first);
    k_logits<<<NE/256, 256, 0, stream>>>(q0, q1, kv0, kv1, dst, scaleA, logits, first);

    fill((float*)segm, 0.0f, NN*8);     // 0 == ordered-encoding minimum
    fill(segs, 0.0f, NN*8);
    fill(z0,   0.0f, NN*8);
    fill(z1,   0.0f, NN*8*3);
    k_segmax<<<(NE*8)/256, 256, 0, stream>>>(logits, dst, segm);
    k_expsum<<<(NE*8)/256, 256, 0, stream>>>(logits, dst, segm, exb, segs);
    k_agg<<<(NE*8)/256, 256, 0, stream>>>(exb, segs, dst, kv0, kv1, z0, z1, first);

    k_proj<<<(NN + 255)/256, 256, 0, stream>>>(f0, f1, z0, z1, Lwp0[L], Lwp1[L], f0n, f1n, first);
    { float* t = f0; f0 = f0n; f0n = t; }
    { float* t = f1; f1 = f1n; f1n = t; }
  }

  // ---------------- final readout ----------------
  k_gather_x0<<<(NE*NC)/256, 256, 0, stream>>>(f0, src, xf0);
  k_make_t<<<NE/256, 256, 0, stream>>>(f1, src, b01, b11, t10, t11);
  fill(hbuf, 0.0f, NE*64);
  runRadial(F00, 2, xf0, hbuf, 64, nullptr);
  runRadial(F10, 2, t10, hbuf, 64, nullptr);
  k_f32tof16<<<(NE*64)/256, 256, 0, stream>>>(hbuf, hb16, NE*64);

  gemm(hb16,  64, mw0f, 2, mb0, NE, 64, 2, 0, hb16b, nullptr, nullptr, nullptr, 0, nullptr);
  gemm(hb16b, 64, mw1f, 2, mb1, NE, 64, 2, 0, hb16,  nullptr, nullptr, nullptr, 0, nullptr);

  fill((float*)d_out, 0.0f, NG);
  k_final<<<NE/256, 256, 0, stream>>>(hb16, mw2, mb2, egraph, escale, (float*)d_out);
}
